// BitLinear_29678224016181
// MI455X (gfx1250) — compile-verified, matrix operand order checked by reading, not run on hardware
//
#include <hip/hip_runtime.h>

typedef __attribute__((ext_vector_type(16))) _Float16 v16h;
typedef __attribute__((ext_vector_type(8)))  _Float16 v8h;
typedef __attribute__((ext_vector_type(8)))  float    v8f;

#define D_IN   2048
#define D_OUT  2048
#define M_ROWS 16384   // B*S = 4*4096
#define EPSV   1e-8f

union FragH {
    v16h v;
    v8h  h[2];
};

// ---------------------------------------------------------------------------
// Kernel 1: per-row RMSNorm of x with affine g, emit fp16 activations.
// One 256-thread block per row of 2048 floats (8 elems/thread, coalesced).
// ---------------------------------------------------------------------------
__global__ __launch_bounds__(256)
void xnorm_kernel(const float* __restrict__ x, const float* __restrict__ g,
                  _Float16* __restrict__ xn) {
    __shared__ float red[256];
    const size_t row = blockIdx.x;
    const float* xr = x + row * D_IN;
    const int tid = threadIdx.x;

    float v[8];
    float ss = 0.f;
#pragma unroll
    for (int i = 0; i < 8; ++i) {
        v[i] = xr[tid + i * 256];
        ss += v[i] * v[i];
    }
    red[tid] = ss;
    __syncthreads();
#pragma unroll
    for (int s = 128; s > 0; s >>= 1) {
        if (tid < s) red[tid] += red[tid + s];
        __syncthreads();
    }
    const float inv = rsqrtf(red[0] * (1.0f / D_IN) + EPSV);

    _Float16* o = xn + row * D_IN;
#pragma unroll
    for (int i = 0; i < 8; ++i) {
        const int j = tid + i * 256;
        o[j] = (_Float16)(v[i] * inv * g[j]);
    }
}

// ---------------------------------------------------------------------------
// Kernel 2: per-row weight prep. Algebra folded to raw w:
//   mask  : |w| > 0.5 * mean|w|
//   scale : (mean|w| / sqrt(mean(w^2)+eps)) * row_scale
//   w_eff : copysign(scale, w) * mask   -> fp16
// ---------------------------------------------------------------------------
__global__ __launch_bounds__(256)
void wprep_kernel(const float* __restrict__ w, const float* __restrict__ row_scale,
                  _Float16* __restrict__ weff) {
    __shared__ float rss[256];
    __shared__ float rsa[256];
    const size_t row = blockIdx.x;
    const float* wr = w + row * D_IN;
    const int tid = threadIdx.x;

    float v[8];
    float ss = 0.f, sa = 0.f;
#pragma unroll
    for (int i = 0; i < 8; ++i) {
        v[i] = wr[tid + i * 256];
        ss += v[i] * v[i];
        sa += fabsf(v[i]);
    }
    rss[tid] = ss;
    rsa[tid] = sa;
    __syncthreads();
#pragma unroll
    for (int s = 128; s > 0; s >>= 1) {
        if (tid < s) { rss[tid] += rss[tid + s]; rsa[tid] += rsa[tid + s]; }
        __syncthreads();
    }
    const float amean = rsa[0] * (1.0f / D_IN);
    const float rinv  = rsqrtf(rss[0] * (1.0f / D_IN) + EPSV);
    const float scale = amean * rinv * row_scale[row];
    const float thr   = 0.5f * amean;

    _Float16* o = weff + row * D_IN;
#pragma unroll
    for (int i = 0; i < 8; ++i) {
        const float wv = v[i];
        const float ov = (fabsf(wv) > thr) ? copysignf(scale, wv) : 0.0f;
        o[tid + i * 256] = (_Float16)ov;
    }
}

// ---------------------------------------------------------------------------
// Kernel 3: WMMA GEMM.  out[M,N] = xn[M,K] * weff[N,K]^T + bias[N]
// Workgroup: 256 threads = 8 wave32s, tile 128(M) x 256(N).
// Wave: 64(M) x 64(N) = 4x4 tiles of 16x16 -> 16 WMMAs per K-step of 32,
// 16 b128 loads per K-step (512 B/WMMA = 32 flop per L2 byte).
// A/B fragments loaded directly from row-major memory per the CDNA5 16-bit
// A-matrix VGPR layout: lane<16 -> K{0..7,16..23}, lane>=16 -> K{8..15,24..31}.
// Both operands are L2-resident (x̂ 64MB + w_eff 8MB << 192MB L2).
// ---------------------------------------------------------------------------
__global__ __launch_bounds__(256)
void gemm_kernel(const _Float16* __restrict__ A, const _Float16* __restrict__ Bw,
                 const float* __restrict__ bias, float* __restrict__ out) {
    const int K = D_IN;
    const int N = D_OUT;

    const int lane = threadIdx.x & 31;
    const int wave = threadIdx.x >> 5;
    const int wm = wave & 1;       // 2 wave slots along M
    const int wn = wave >> 1;      // 4 wave slots along N
    const int lrow = lane & 15;    // row/col within 16x16 tile
    const int hi   = lane >> 4;    // half-wave select (K interleave / M offset)

    const int mBase = blockIdx.y * 128 + wm * 64;
    const int nBase = blockIdx.x * 256 + wn * 64;

    const _Float16* aPtr[4];
#pragma unroll
    for (int mi = 0; mi < 4; ++mi)
        aPtr[mi] = A + (size_t)(mBase + mi * 16 + lrow) * K + hi * 8;

    const _Float16* bPtr[4];
#pragma unroll
    for (int ni = 0; ni < 4; ++ni)
        bPtr[ni] = Bw + (size_t)(nBase + ni * 16 + lrow) * K + hi * 8;

    v8f acc[4][4];
#pragma unroll
    for (int mi = 0; mi < 4; ++mi)
#pragma unroll
        for (int ni = 0; ni < 4; ++ni)
            acc[mi][ni] = (v8f){0.f, 0.f, 0.f, 0.f, 0.f, 0.f, 0.f, 0.f};

    for (int k0 = 0; k0 < K; k0 += 32) {
        // Unconditional speculative prefetch ~8 K-steps ahead on the streamed
        // A operand (global_prefetch_b8; OOB translations dropped silently).
        __builtin_prefetch(aPtr[0] + k0 + 256, 0, 3);
        __builtin_prefetch(aPtr[2] + k0 + 256, 0, 3);

        FragH a[4];
#pragma unroll
        for (int mi = 0; mi < 4; ++mi) {
            a[mi].h[0] = *(const v8h*)(aPtr[mi] + k0);        // K lo 8 (interleaved)
            a[mi].h[1] = *(const v8h*)(aPtr[mi] + k0 + 16);   // K hi 8
        }
        FragH b[4];
#pragma unroll
        for (int ni = 0; ni < 4; ++ni) {
            b[ni].h[0] = *(const v8h*)(bPtr[ni] + k0);
            b[ni].h[1] = *(const v8h*)(bPtr[ni] + k0 + 16);
        }

#pragma unroll
        for (int mi = 0; mi < 4; ++mi)
#pragma unroll
            for (int ni = 0; ni < 4; ++ni)
                acc[mi][ni] = __builtin_amdgcn_wmma_f32_16x16x32_f16(
                    /*neg_a=*/false, a[mi].v, /*neg_b=*/false, b[ni].v,
                    /*c_mod=*/(short)0, acc[mi][ni],
                    /*reuse_a=*/false, /*reuse_b=*/false);
    }

    // Epilogue: C/D layout -> element (m,n): n = lane&15 (per n-tile),
    // m = tileM + vgpr + 8*(lane>>4). Add bias[n] (column-constant).
#pragma unroll
    for (int ni = 0; ni < 4; ++ni) {
        const int n = nBase + ni * 16 + lrow;
        const float bv = bias[n];
#pragma unroll
        for (int mi = 0; mi < 4; ++mi) {
            const int mrow = mBase + mi * 16 + hi * 8;
#pragma unroll
            for (int r = 0; r < 8; ++r)
                out[(size_t)(mrow + r) * N + n] = acc[mi][ni][r] + bv;
        }
    }
}

// ---------------------------------------------------------------------------
extern "C" void kernel_launch(void* const* d_in, const int* in_sizes, int n_in,
                              void* d_out, int out_size, void* d_ws, size_t ws_size,
                              hipStream_t stream) {
    const float* x         = (const float*)d_in[0];   // [4,4096,2048]
    const float* weight    = (const float*)d_in[1];   // [2048,2048]
    const float* row_scale = (const float*)d_in[2];   // [2048,1]
    const float* bias      = (const float*)d_in[3];   // [2048]
    const float* g         = (const float*)d_in[4];   // [2048]
    float* out             = (float*)d_out;           // [4,4096,2048] f32

    // Workspace layout: fp16 activations (64MB) then fp16 effective weights (8MB)
    _Float16* xn   = (_Float16*)d_ws;
    _Float16* weff = xn + (size_t)M_ROWS * D_IN;

    xnorm_kernel<<<M_ROWS, 256, 0, stream>>>(x, g, xn);
    wprep_kernel<<<D_OUT, 256, 0, stream>>>(weight, row_scale, weff);

    dim3 grid(D_OUT / 256, M_ROWS / 128);   // (8, 128)
    gemm_kernel<<<grid, dim3(256), 0, stream>>>(xn, weff, bias, out);
}
